// MambaTest_27754078666825
// MI455X (gfx1250) — matrix operand order repaired and checked by reading, not compiled, not run on hardware
//
#include <hip/hip_runtime.h>
#include <hip/hip_bf16.h>

// ---------------------------------------------------------------------------
// MI455X / gfx1250 implementation: Mamba encoder + 2-layer bi-GRU.
// All dense projections run through v_wmma_f32_16x16x32_bf16 (fp32 accum).
// Sequential scans (SSM, GRU) stay VALU-resident: LDS-staged broadcast data,
// wave32-lockstep recurrences.
// ---------------------------------------------------------------------------

typedef __bf16 bf16;
typedef __attribute__((ext_vector_type(16))) __bf16 v16bf;
typedef __attribute__((ext_vector_type(8)))  __bf16 v8bf;
typedef __attribute__((ext_vector_type(8)))  float  v8f;

#define BB   256
#define LL   50
#define DM   512
#define DI   4096
#define DSNT 21
#define DTR  32
#define HG   25
#define MROWS (BB*LL)          // 12800

// ---- WMMA fragment loaders (contiguous-K layout, wave32) -------------------
// A: 16x32 bf16, row-major source [16 x K]. lane&15 = M row; each lane
//    holds 16 contiguous K ((lane>>4)*16 band), matching load_b_frag.
__device__ __forceinline__ v16bf load_a_frag(const bf16* __restrict__ p0, int ld) {
  const int lane = threadIdx.x & 31;
  const bf16* p = p0 + (long)(lane & 15) * ld + ((lane >> 4) << 4);
  v8bf lo = *(const v8bf*)p;
  v8bf hi = *(const v8bf*)(p + 8);
  v16bf f;
#pragma unroll
  for (int i = 0; i < 8; ++i) { f[i] = lo[i]; f[i + 8] = hi[i]; }
  return f;
}
// B: 32x16 bf16 from row-major weights W[N][K]. lane&15 = N col; each lane
// holds 16 contiguous K ((lane>>4)*16 band) -> one 32B load.
__device__ __forceinline__ v16bf load_b_frag(const bf16* __restrict__ p0, int ld) {
  const int lane = threadIdx.x & 31;
  const bf16* p = p0 + (long)(lane & 15) * ld + ((lane >> 4) << 4);
  return *(const v16bf*)p;
}

// ---- Generic WMMA GEMM: C[M x N] = A[M x K] * B[N x K]^T ------------------
// Block = 8 waves; wave w owns 16x16 tile at (blockIdx.x*128 + w*16, blockIdx.y*16).
// epi: 0 = plain fp32 store, 1 = softplus(acc + bias[n]) (dt path).
__global__ __launch_bounds__(256) void wmma_gemm_kernel(
    const bf16* __restrict__ A, const bf16* __restrict__ Bw,
    float* __restrict__ C, const float* __restrict__ bias,
    int M, int K, int ldc, int epi)
{
  const int wave = threadIdx.x >> 5;
  const int lane = threadIdx.x & 31;
  const int m0 = blockIdx.x * 128 + wave * 16;
  const int n0 = blockIdx.y * 16;
  if (m0 >= M) return;
  const bf16* Ap = A + (long)m0 * K;
  const bf16* Bp = Bw + (long)n0 * K;
  v8f acc = {};
  for (int k = 0; k < K; k += 32) {
    v16bf a = load_a_frag(Ap + k, K);
    v16bf b = load_b_frag(Bp + k, K);
    acc = __builtin_amdgcn_wmma_f32_16x16x32_bf16(
        false, a, false, b, (short)0, acc, false, false);
  }
  const int rbase = m0 + ((lane >> 4) << 3);
  const int col   = n0 + (lane & 15);
#pragma unroll
  for (int r = 0; r < 8; ++r) {
    float v = acc[r];
    if (epi == 1) {
      v += bias[col];
      v = (v > 20.f) ? v : log1pf(__expf(v));   // softplus
    }
    C[(long)(rbase + r) * ldc + col] = v;
  }
}

// ---- Embedding gather + bf16 cast -----------------------------------------
__global__ void embed_kernel(const int* __restrict__ oe,
                             const float* __restrict__ emb,
                             bf16* __restrict__ X, int total)
{
  int i = blockIdx.x * 256 + threadIdx.x;
  if (i >= total) return;
  int r = i >> 9, c = i & (DM - 1);
  X[i] = (bf16)emb[(long)oe[r] * DM + c];
}

// ---- fp32 -> bf16 cast with zero padding (weights / activations) ----------
__global__ void cast_pad_kernel(const float* __restrict__ src, int sld,
                                int srows, int scols,
                                bf16* __restrict__ dst, int dld, int drows)
{
  long i = (long)blockIdx.x * 256 + threadIdx.x;
  long total = (long)drows * dld;
  if (i >= total) return;
  int r = (int)(i / dld), c = (int)(i % dld);
  float v = (r < srows && c < scols) ? src[(long)r * sld + c] : 0.f;
  dst[i] = (bf16)v;
}

// ---- depthwise conv (k=2) + SiLU on x-half; SiLU(z) in place on z-half ----
__global__ void conv_silu_kernel(float* __restrict__ xz,
                                 const float* __restrict__ cw,
                                 const float* __restrict__ cb,
                                 float* __restrict__ xc_f,
                                 bf16* __restrict__ xc_b)
{
  long i = (long)blockIdx.x * 256 + threadIdx.x;
  if (i >= (long)MROWS * DI) return;
  int row = (int)(i >> 12);           // /4096
  int d   = (int)(i & (DI - 1));
  int l   = row % LL;
  float xv = xz[(long)row * (2 * DI) + d];
  float xp = (l > 0) ? xz[(long)(row - 1) * (2 * DI) + d] : 0.f;
  float v = xp * cw[d * 2] + xv * cw[d * 2 + 1] + cb[d];
  v = v / (1.f + __expf(-v));                       // SiLU
  xc_f[i] = v;
  xc_b[i] = (bf16)v;
  float z = xz[(long)row * (2 * DI) + DI + d];
  xz[(long)row * (2 * DI) + DI + d] = z / (1.f + __expf(-z));
}

// ---- selective-scan: 1 thread per (b,d); B_t/C_t staged in LDS ------------
__global__ __launch_bounds__(256) void scan_kernel(
    const float* __restrict__ xdbl, const float* __restrict__ dt,
    const float* __restrict__ xc,   const float* __restrict__ xz,
    const float* __restrict__ A_log, const float* __restrict__ Dp,
    bf16* __restrict__ y_bf)
{
  __shared__ float sB[LL * DSNT];
  __shared__ float sC[LL * DSNT];
  const int b = blockIdx.y;
  const int d = blockIdx.x * 256 + threadIdx.x;
  for (int i = threadIdx.x; i < LL * DSNT; i += 256) {
    int t = i / DSNT, s = i % DSNT;
    const float* row = xdbl + (long)(b * LL + t) * 80;
    sB[i] = row[DTR + s];            // cols 32..52
    sC[i] = row[DTR + DSNT + s];     // cols 53..73
  }
  __syncthreads();
  float a[DSNT], h[DSNT];
#pragma unroll
  for (int s = 0; s < DSNT; ++s) { a[s] = -__expf(A_log[(long)d * DSNT + s]); h[s] = 0.f; }
  const float Dd = Dp[d];
  for (int t = 0; t < LL; ++t) {
    long row = (long)(b * LL + t);
    float dtv = dt[row * DI + d];
    float xv  = xc[row * DI + d];
    float bx  = dtv * xv;
    float acc = 0.f;
#pragma unroll
    for (int s = 0; s < DSNT; ++s) {
      h[s] = h[s] * __expf(dtv * a[s]) + bx * sB[t * DSNT + s];
      acc += h[s] * sC[t * DSNT + s];
    }
    float y = acc + xv * Dd;
    y *= xz[row * (2 * DI) + DI + d];                 // * SiLU(z)
    y_bf[row * DI + d] = (bf16)y;
  }
}

// ---- GRU direction scan: 1 wave32 per batch element -----------------------
// gx already holds x@Wih^T (padded N=80); bih added here. Whh rows register-
// resident per lane; h lives in LDS (wave-lockstep recurrence).
__global__ __launch_bounds__(256) void gru_kernel(
    const float* __restrict__ gx,  const float* __restrict__ bih,
    const float* __restrict__ Whh, const float* __restrict__ bhh,
    float* __restrict__ outp, int off, int rev)
{
  __shared__ float sh[8][32];
  const int w = threadIdx.x >> 5;
  const int j = threadIdx.x & 31;
  const int b = blockIdx.x * 8 + w;
  sh[w][j] = 0.f;
  float wr[HG], wz[HG], wn[HG];
  float br = 0, bz = 0, bn = 0, bir = 0, biz = 0, bin_ = 0;
  if (j < HG) {
#pragma unroll
    for (int k = 0; k < HG; ++k) {
      wr[k] = Whh[(j) * HG + k];
      wz[k] = Whh[(HG + j) * HG + k];
      wn[k] = Whh[(2 * HG + j) * HG + k];
    }
    br = bhh[j]; bz = bhh[HG + j]; bn = bhh[2 * HG + j];
    bir = bih[j]; biz = bih[HG + j]; bin_ = bih[2 * HG + j];
  }
  __syncthreads();
  for (int t = 0; t < LL; ++t) {
    int tt = rev ? (LL - 1 - t) : t;
    long row = (long)b * LL + tt;
    float hnew = 0.f;
    if (j < HG) {
      float hr = br, hz = bz, hn = bn;
#pragma unroll
      for (int k = 0; k < HG; ++k) {
        float hk = sh[w][k];
        hr += wr[k] * hk; hz += wz[k] * hk; hn += wn[k] * hk;
      }
      float xr  = gx[row * 80 + j]          + bir + hr;
      float xzv = gx[row * 80 + HG + j]     + biz + hz;
      float xn  = gx[row * 80 + 2 * HG + j] + bin_;
      float r   = 1.f / (1.f + __expf(-xr));
      float zz  = 1.f / (1.f + __expf(-xzv));
      float n   = tanhf(xn + r * hn);
      hnew = (1.f - zz) * n + zz * sh[w][j];
      outp[row * (2 * HG) + off + j] = hnew;
    }
    __syncthreads();
    if (j < HG) sh[w][j] = hnew;
    __syncthreads();
  }
}

// ---------------------------------------------------------------------------
extern "C" void kernel_launch(void* const* d_in, const int* in_sizes, int n_in,
                              void* d_out, int out_size, void* d_ws, size_t ws_size,
                              hipStream_t stream) {
  (void)in_sizes; (void)n_in; (void)out_size; (void)ws_size;
  const int*   oe     = (const int*)  d_in[0];
  const float* emb    = (const float*)d_in[1];
  const float* W_in   = (const float*)d_in[2];
  const float* conv_w = (const float*)d_in[3];
  const float* conv_b = (const float*)d_in[4];
  const float* W_xprj = (const float*)d_in[5];
  const float* W_dt   = (const float*)d_in[6];
  const float* b_dt   = (const float*)d_in[7];
  const float* A_log  = (const float*)d_in[8];
  const float* Dvec   = (const float*)d_in[9];
  const float* W_out  = (const float*)d_in[10];
  const float* Wih0f = (const float*)d_in[11]; const float* Whh0f = (const float*)d_in[12];
  const float* bih0f = (const float*)d_in[13]; const float* bhh0f = (const float*)d_in[14];
  const float* Wih0b = (const float*)d_in[15]; const float* Whh0b = (const float*)d_in[16];
  const float* bih0b = (const float*)d_in[17]; const float* bhh0b = (const float*)d_in[18];
  const float* Wih1f = (const float*)d_in[19]; const float* Whh1f = (const float*)d_in[20];
  const float* bih1f = (const float*)d_in[21]; const float* bhh1f = (const float*)d_in[22];
  const float* Wih1b = (const float*)d_in[23]; const float* Whh1b = (const float*)d_in[24];
  const float* bih1b = (const float*)d_in[25]; const float* bhh1b = (const float*)d_in[26];

  char* ws = (char*)d_ws;
  size_t off = 0;
  auto take = [&](size_t bytes) -> void* {
    void* p = ws + off;
    off = (off + bytes + 255) & ~(size_t)255;
    return p;
  };
  const int M = MROWS;
  bf16*  Xbf    = (bf16*) take((size_t)M * DM * 2);
  bf16*  Winbf  = (bf16*) take((size_t)2 * DI * DM * 2);
  bf16*  Wxpbf  = (bf16*) take((size_t)80 * DI * 2);
  bf16*  Wdtbf  = (bf16*) take((size_t)DI * DTR * 2);
  bf16*  Woutbf = (bf16*) take((size_t)DM * DI * 2);
  bf16*  W0fbf  = (bf16*) take((size_t)80 * DM * 2);
  bf16*  W0bbf  = (bf16*) take((size_t)80 * DM * 2);
  bf16*  W1fbf  = (bf16*) take((size_t)80 * 64 * 2);
  bf16*  W1bbf  = (bf16*) take((size_t)80 * 64 * 2);
  float* xz     = (float*)take((size_t)M * 2 * DI * 4);
  float* xc_f   = (float*)take((size_t)M * DI * 4);
  bf16*  xc_b   = (bf16*) take((size_t)M * DI * 2);
  float* xdbl   = (float*)take((size_t)M * 80 * 4);
  bf16*  dtlo   = (bf16*) take((size_t)M * DTR * 2);
  float* dtbuf  = (float*)take((size_t)M * DI * 4);
  bf16*  y_bf   = (bf16*) take((size_t)M * DI * 2);
  float* enc    = (float*)take((size_t)M * DM * 4);
  bf16*  encbf  = (bf16*) take((size_t)M * DM * 2);
  float* gx0f   = (float*)take((size_t)M * 80 * 4);
  float* gx0b   = (float*)take((size_t)M * 80 * 4);
  float* h1     = (float*)take((size_t)M * 50 * 4);
  bf16*  h1p    = (bf16*) take((size_t)M * 64 * 2);
  float* gx1f   = (float*)take((size_t)M * 80 * 4);
  float* gx1b   = (float*)take((size_t)M * 80 * 4);

  dim3 blk(256);
  auto g1 = [](long n) { return dim3((unsigned)((n + 255) / 256)); };

  // 1) embedding gather -> bf16
  embed_kernel<<<g1((long)M * DM), blk, 0, stream>>>(oe, emb, Xbf, M * DM);

  // 2) weight casts / zero-padded staging
  cast_pad_kernel<<<g1((long)2 * DI * DM), blk, 0, stream>>>(W_in,  DM,  2*DI, DM,  Winbf, DM,  2*DI);
  cast_pad_kernel<<<g1((long)80 * DI),     blk, 0, stream>>>(W_xprj,DI,  74,   DI,  Wxpbf, DI,  80);
  cast_pad_kernel<<<g1((long)DI * DTR),    blk, 0, stream>>>(W_dt,  DTR, DI,   DTR, Wdtbf, DTR, DI);
  cast_pad_kernel<<<g1((long)DM * DI),     blk, 0, stream>>>(W_out, DI,  DM,   DI,  Woutbf,DI,  DM);
  cast_pad_kernel<<<g1((long)80 * DM),     blk, 0, stream>>>(Wih0f, DM,  75,   DM,  W0fbf, DM,  80);
  cast_pad_kernel<<<g1((long)80 * DM),     blk, 0, stream>>>(Wih0b, DM,  75,   DM,  W0bbf, DM,  80);
  cast_pad_kernel<<<g1((long)80 * 64),     blk, 0, stream>>>(Wih1f, 50,  75,   50,  W1fbf, 64,  80);
  cast_pad_kernel<<<g1((long)80 * 64),     blk, 0, stream>>>(Wih1b, 50,  75,   50,  W1bbf, 64,  80);

  // 3) xz = X @ W_in^T   (12800 x 8192, K=512)
  wmma_gemm_kernel<<<dim3(M/128, (2*DI)/16), blk, 0, stream>>>(Xbf, Winbf, xz, nullptr, M, DM, 2*DI, 0);

  // 4) depthwise conv + SiLU; SiLU(z) in place
  conv_silu_kernel<<<g1((long)M * DI), blk, 0, stream>>>(xz, conv_w, conv_b, xc_f, xc_b);

  // 5) xdbl = xc @ W_xproj^T   (N padded 74->80, K=4096)
  wmma_gemm_kernel<<<dim3(M/128, 80/16), blk, 0, stream>>>(xc_b, Wxpbf, xdbl, nullptr, M, DI, 80, 0);

  // 6) dt_lo -> bf16; dt = softplus(dt_lo @ W_dt^T + b_dt)  (K=32)
  cast_pad_kernel<<<g1((long)M * DTR), blk, 0, stream>>>(xdbl, 80, M, DTR, dtlo, DTR, M);
  wmma_gemm_kernel<<<dim3(M/128, DI/16), blk, 0, stream>>>(dtlo, Wdtbf, dtbuf, b_dt, M, DTR, DI, 1);

  // 7) selective scan -> y (bf16, already *SiLU(z) and +x*D)
  scan_kernel<<<dim3(DI/256, BB), blk, 0, stream>>>(xdbl, dtbuf, xc_f, xz, A_log, Dvec, y_bf);

  // 8) enc = y @ W_out^T  (K=4096) ; cast to bf16
  wmma_gemm_kernel<<<dim3(M/128, DM/16), blk, 0, stream>>>(y_bf, Woutbf, enc, nullptr, M, DI, DM, 0);
  cast_pad_kernel<<<g1((long)M * DM), blk, 0, stream>>>(enc, DM, M, DM, encbf, DM, M);

  // 9) GRU layer 0: batched input projections, then wave-per-batch scans
  wmma_gemm_kernel<<<dim3(M/128, 80/16), blk, 0, stream>>>(encbf, W0fbf, gx0f, nullptr, M, DM, 80, 0);
  wmma_gemm_kernel<<<dim3(M/128, 80/16), blk, 0, stream>>>(encbf, W0bbf, gx0b, nullptr, M, DM, 80, 0);
  gru_kernel<<<dim3(BB/8), blk, 0, stream>>>(gx0f, bih0f, Whh0f, bhh0f, h1, 0,  0);
  gru_kernel<<<dim3(BB/8), blk, 0, stream>>>(gx0b, bih0b, Whh0b, bhh0b, h1, HG, 1);

  // 10) GRU layer 1 (input dim 50 padded to 64); writes d_out directly
  cast_pad_kernel<<<g1((long)M * 64), blk, 0, stream>>>(h1, 50, M, 50, h1p, 64, M);
  wmma_gemm_kernel<<<dim3(M/128, 80/16), blk, 0, stream>>>(h1p, W1fbf, gx1f, nullptr, M, 64, 80, 0);
  wmma_gemm_kernel<<<dim3(M/128, 80/16), blk, 0, stream>>>(h1p, W1bbf, gx1b, nullptr, M, 64, 80, 0);
  gru_kernel<<<dim3(BB/8), blk, 0, stream>>>(gx1f, bih1f, Whh1f, bhh1f, (float*)d_out, 0,  0);
  gru_kernel<<<dim3(BB/8), blk, 0, stream>>>(gx1b, bih1b, Whh1b, bhh1b, (float*)d_out, HG, 1);
}
